// Geo_ALIF_SNN_90512140796398
// MI455X (gfx1250) — matrix operand order repaired and checked by reading, not compiled, not run on hardware
//
#include <hip/hip_runtime.h>

typedef __attribute__((ext_vector_type(16))) _Float16 v16h;
typedef __attribute__((ext_vector_type(4)))  _Float16 h4;
typedef __attribute__((ext_vector_type(8)))  float    v8f;
typedef __attribute__((ext_vector_type(4)))  float    f32x4;
typedef __attribute__((ext_vector_type(4)))  unsigned int u32x4;
typedef __attribute__((ext_vector_type(8)))  int      i32x8;
typedef __attribute__((ext_vector_type(4)))  int      i32x4;

#if defined(__HIP_DEVICE_COMPILE__) && __has_builtin(__builtin_amdgcn_tensor_load_to_lds) && \
    __has_builtin(__builtin_amdgcn_s_wait_tensorcnt)
#define GEO_HAS_TDM 1
#else
#define GEO_HAS_TDM 0
#endif

namespace {
constexpr int Bt = 4096, Tt = 320, CE = 64, CG = 18, C1 = 32, C2 = 16;
constexpr int BPB = 16;        // batch rows per block
constexpr int NTHREADS = 256;  // 8 wave32 waves
constexpr int CHK = 16;        // timesteps per chunk
constexpr int NCH = Tt / CHK;  // 20 chunks

constexpr float THETA0 = 0.5f, BETA = 1.8f, TAU_A = 0.36f, TAU_M = 0.8f, EPSB = 1e-5f;

constexpr int FRAG_H = 512;             // one swizzled 32x16 B fragment: 32 lanes x 16 halves

// LDS carve (halves first, then floats)
constexpr int XE_H   = BPB * 20 * CE;   // x_eeg f16 operand buffer: 16 rows + 4 halo
constexpr int RING_H = BPB * 48 * C1;   // conv1 output ring (3 chunks deep)
constexpr int XG_H   = BPB * CHK * 32;  // x_geo staging, K padded 18->32
constexpr int WK1_H  = 20 * FRAG_H;     // conv1 weights: (w,ks,nt) fragments
constexpr int WK2_H  = 3 * FRAG_H;      // conv2 weights: (w) fragments
constexpr int GW_H   = 1 * FRAG_H;      // geo_W fragment (K padded 18->32)
constexpr int W1_H   = 20 * FRAG_H;     // fc1 weights: (ks,nt) fragments
constexpr int W2_H   = 2 * FRAG_H;      // fc2 weights: (ks) fragments, N padded 4->16
constexpr int DP_H   = BPB * 160;
constexpr int H1_H   = BPB * 64;
constexpr int HALF_TOTAL = XE_H + RING_H + XG_H + WK1_H + WK2_H + GW_H + W1_H + W2_H + DP_H + H1_H;

constexpr int XEF32_F = BPB * 20 * CE;  // TDM landing buffer (raw f32), 80 KB
constexpr int ENC_F = BPB * C2 * CHK;   // [b][h][t] (transposed for scan reads)
constexpr int GEO_F = BPB * C2 * CHK;
constexpr int FLOAT_TOTAL = XEF32_F + ENC_F + GEO_F + 32 + 32 + 16 + 16 + 64 + 16;

constexpr unsigned XEF32_BYTE_OFF = unsigned(HALF_TOTAL) * 2u;  // dyn-LDS base is offset 0
constexpr size_t SMEM_BYTES = size_t(HALF_TOTAL) * 2 + size_t(FLOAT_TOTAL) * 4;
} // namespace

// --- WMMA fragment loaders (CDNA5 16-bit layouts, CDNA5 ISA 7.12.2) ---
__device__ __forceinline__ v16h load_a16(const _Float16* base, int ld, int kofs, int lane) {
  const int m = lane & 15, g = lane >> 4;
  const _Float16* row = base + m * ld + kofs;
  v16h a;
#pragma unroll
  for (int i = 0; i < 8; ++i) {
    const int kb = ((i & 4) ? 16 : 0) + g * 8 + (i & 3) * 2;
    a[2 * i]     = row[kb];
    a[2 * i + 1] = row[kb + 1];
  }
  return a;
}
__device__ __forceinline__ v16h load_bfrag(const _Float16* frag, int lane) {
  return *reinterpret_cast<const v16h*>(frag + lane * 16);
}
// A fragment from the 48-row ring; rbase = (row of m=0) already reduced to [-1, 34]
__device__ __forceinline__ v16h load_a_ring(const _Float16* ring_b, int rbase, int lane) {
  const int m = lane & 15, g = lane >> 4;
  int r = rbase + m;                 // [-1, 49]
  r += (r < 0) ? 48 : 0;
  r -= (r >= 48) ? 48 : 0;
  const _Float16* row = ring_b + r * C1;
  v16h a;
#pragma unroll
  for (int i = 0; i < 8; ++i) {
    const int kb = ((i & 4) ? 16 : 0) + g * 8 + (i & 3) * 2;
    a[2 * i]     = row[kb];
    a[2 * i + 1] = row[kb + 1];
  }
  return a;
}

__device__ __forceinline__ void stage_bfrag(_Float16* dst, const float* __restrict__ W,
                                            int ld, int kofs, int nofs, int kmax, int nmax,
                                            int tid) {
  for (int e = tid; e < FRAG_H; e += NTHREADS) {
    const int l = e >> 4, j = e & 15;
    const int k = kofs + ((j & 8) << 1) + ((l & 16) >> 1) + (j & 7);
    const int n = nofs + (l & 15);
    float v = 0.f;
    if (k < kmax && n < nmax) v = W[k * ld + n];
    dst[e] = (_Float16)v;
  }
}

// ---- Tensor Data Mover: one 3D descriptor moves the whole 16b x 20t x 64ch f32 tile ----
__device__ __forceinline__ void tdm_issue_xeeg(const float* gsrc, int valid_rows) {
#if GEO_HAS_TDM
  const unsigned long long ga = (unsigned long long)(const void*)gsrc;
  u32x4 g0;
  g0[0] = 1u;                                            // count=1, user D#, no gather
  g0[1] = XEF32_BYTE_OFF;                                // lds_addr (bytes)
  g0[2] = (unsigned)(ga & 0xFFFFFFFFull);                // global_addr[31:0]
  g0[3] = (unsigned)((ga >> 32) & 0x01FFFFFFull) | (2u << 30);  // addr[56:32] | type=2
  i32x8 g1;
  g1[0] = (int)(2u << 16);                               // data_size=4B; mask=0; no pad/iter
  g1[1] = (int)((64u & 0xFFFFu) << 16);                  // tensor_dim0 lo16 (=64)
  g1[2] = (int)((64u >> 16) | (((unsigned)valid_rows & 0xFFFFu) << 16)); // dim0 hi | dim1 lo
  g1[3] = (int)((((unsigned)valid_rows >> 16) & 0xFFFFu) | (64u << 16)); // dim1 hi | tile_dim0=64
  g1[4] = (int)(20u | (16u << 16));                      // tile_dim1=20, tile_dim2=16
  g1[5] = (int)64u;                                      // tensor_dim0_stride lo32 (=row, 64)
  g1[6] = (int)(((unsigned)(Tt * CE) & 0xFFFFu) << 16);  // dim0_stride hi=0 | dim1_stride lo16
  g1[7] = (int)((unsigned)(Tt * CE) >> 16);              // tensor_dim1_stride hi32 (batch stride)
  i32x4 g2;
  g2[0] = 16;                                            // tensor_dim2 (batch) = 16
  g2[1] = 0; g2[2] = 0; g2[3] = 0;                       // no dim3 / tile_dim3=0
  i32x4 g3 = {0, 0, 0, 0};
#if __clang_major__ >= 23
  i32x8 g4 = {0, 0, 0, 0, 0, 0, 0, 0};
  __builtin_amdgcn_tensor_load_to_lds(g0, g1, g2, g3, g4, 0);
#else
  __builtin_amdgcn_tensor_load_to_lds(g0, g1, g2, g3, 0);
#endif
#else
  (void)gsrc; (void)valid_rows;
#endif
}
__device__ __forceinline__ void tdm_wait() {
#if GEO_HAS_TDM
  __builtin_amdgcn_s_wait_tensorcnt(0);
#endif
}

__global__ __launch_bounds__(NTHREADS, 1)
void geo_alif_fused(const float* __restrict__ x_eeg, const float* __restrict__ x_geo,
                    const float* __restrict__ k1,    const float* __restrict__ cb1,
                    const float* __restrict__ bn1s,  const float* __restrict__ bn1b,
                    const float* __restrict__ bn1m,  const float* __restrict__ bn1v,
                    const float* __restrict__ k2,    const float* __restrict__ cb2,
                    const float* __restrict__ bn2s,  const float* __restrict__ bn2b,
                    const float* __restrict__ bn2m,  const float* __restrict__ bn2v,
                    const float* __restrict__ geoW,  const float* __restrict__ geob,
                    const float* __restrict__ gamma, const float* __restrict__ w1,
                    const float* __restrict__ b1,    const float* __restrict__ w2,
                    const float* __restrict__ b2,    float* __restrict__ out) {
#if GEO_HAS_TDM
  constexpr bool kTDM = true;
#else
  constexpr bool kTDM = false;
#endif
  extern __shared__ __align__(32) char smraw[];
  _Float16* hp = reinterpret_cast<_Float16*>(smraw);
  _Float16* sm_xe   = hp; hp += XE_H;
  _Float16* sm_ring = hp; hp += RING_H;
  _Float16* sm_xg   = hp; hp += XG_H;
  _Float16* sm_wk1  = hp; hp += WK1_H;
  _Float16* sm_wk2  = hp; hp += WK2_H;
  _Float16* sm_gw   = hp; hp += GW_H;
  _Float16* sm_w1   = hp; hp += W1_H;
  _Float16* sm_w2   = hp; hp += W2_H;
  _Float16* sm_dp   = hp; hp += DP_H;
  _Float16* sm_h1   = hp; hp += H1_H;
  float* fp = reinterpret_cast<float*>(smraw + size_t(HALF_TOTAL) * 2);
  float* sm_xef32 = fp; fp += XEF32_F;   // 16-byte aligned (HALF_TOTAL*2 % 16 == 0)
  float* sm_enc  = fp; fp += ENC_F;
  float* sm_geo  = fp; fp += GEO_F;
  float* sm_bn1A = fp; fp += 32;
  float* sm_bn1B = fp; fp += 32;
  float* sm_bn2A = fp; fp += 16;
  float* sm_bn2B = fp; fp += 16;
  float* sm_fc1b = fp; fp += 64;
  float* sm_geob = fp; fp += 16;

  const int tid  = threadIdx.x;
  const int wid  = tid >> 5;
  const int lane = tid & 31;
  const int b0   = blockIdx.x * BPB;

  // ---- one-time staging: weights swizzled into B-fragment layout, BN folds, ring clear ----
#pragma unroll
  for (int w = 0; w < 5; ++w)
#pragma unroll
    for (int ks = 0; ks < 2; ++ks)
#pragma unroll
      for (int nt = 0; nt < 2; ++nt)
        stage_bfrag(sm_wk1 + (((w * 2 + ks) * 2) + nt) * FRAG_H,
                    k1 + w * (CE * C1), C1, ks * 32, nt * 16, CE, C1, tid);
#pragma unroll
  for (int w = 0; w < 3; ++w)
    stage_bfrag(sm_wk2 + w * FRAG_H, k2 + w * (C1 * C2), C2, 0, 0, C1, C2, tid);
  stage_bfrag(sm_gw, geoW, C2, 0, 0, CG, C2, tid);
#pragma unroll
  for (int ks = 0; ks < 5; ++ks)
#pragma unroll
    for (int nt = 0; nt < 4; ++nt)
      stage_bfrag(sm_w1 + (ks * 4 + nt) * FRAG_H, w1, 64, ks * 32, nt * 16, 160, 64, tid);
#pragma unroll
  for (int ks = 0; ks < 2; ++ks)
    stage_bfrag(sm_w2 + ks * FRAG_H, w2, 4, ks * 32, 0, 64, 4, tid);

  for (int f = tid; f < RING_H; f += NTHREADS) sm_ring[f] = (_Float16)0.f;
  if (tid < 32) {
    const float rs = rsqrtf(bn1v[tid] + EPSB);
    const float A  = rs * bn1s[tid];
    sm_bn1A[tid] = A;
    sm_bn1B[tid] = (cb1[tid] - bn1m[tid]) * A + bn1b[tid];
  }
  if (tid < 16) {
    const float rs = rsqrtf(bn2v[tid] + EPSB);
    const float A  = rs * bn2s[tid];
    sm_bn2A[tid] = A;
    sm_bn2B[tid] = (cb2[tid] - bn2m[tid]) * A + bn2b[tid];
    sm_geob[tid] = geob[tid];
  }
  if (tid < 64) sm_fc1b[tid] = b1[tid];

  // ---- ALIF scan state: thread = (batch row sb, hidden unit sh) ----
  const int sb = tid >> 4, sh = tid & 15;
  const float gam = gamma[sh];
  float eta = 0.f, mem = 0.f, li = 0.f, spk = 0.f, s0 = 0.f, s1 = 0.f;

  int baseA = 0;                      // tA % 48, tracked incrementally (tA step 16, ring 48)

  for (int c = 0; c <= NCH; ++c) {
    const int tA = c * CHK;           // conv1 output rows this chunk
    const int tB = tA - CHK;          // conv2/geo/scan rows (lag one chunk)
    const int baseB = (baseA >= CHK) ? (baseA - CHK) : (baseA + 32);  // tB % 48

    // --- arrival phase: fallback stage (chunk 0 / no-TDM) or wait for TDM ---
    if (c < NCH && (!kTDM || c == 0)) {
      const int b = tid >> 4, j = tid & 15;
      for (int rr = j; rr < 320; rr += 16) {    // 320 float4-quads per batch row
        const int r = rr >> 4, c4 = rr & 15;
        const int t = tA - 2 + r;
        f32x4 v = {0.f, 0.f, 0.f, 0.f};
        if ((unsigned)t < (unsigned)Tt)
          v = *reinterpret_cast<const f32x4*>(x_eeg + (size_t)((b0 + b) * Tt + t) * CE + c4 * 4);
        *reinterpret_cast<f32x4*>(sm_xef32 + (b * 320 + rr) * 4) = v;
      }
    }
    if (kTDM && c >= 1 && c < NCH && wid == 0) tdm_wait();
    if (c >= 1) {
      for (int f = tid; f < XG_H; f += NTHREADS) {
        const int b = f >> 9, r = (f >> 5) & 15, ch = f & 31;
        float v = 0.f;
        if (ch < CG) v = x_geo[(size_t)((b0 + b) * Tt + (tB + r)) * CG + ch];
        sm_xg[f] = (_Float16)v;
      }
      if (tB + 2 * CHK < Tt)   // prefetch next x_geo chunk (global_prefetch_b8)
        __builtin_prefetch(x_geo + (size_t)((b0 + sb) * Tt + (tB + CHK + sh)) * CG, 0, 0);
    }
    if (c == NCH) {  // conv2 at t=319 needs conv1[320] == 0: clear ring slot 320%48==32
      for (int f = tid; f < BPB * C1; f += NTHREADS) {
        const int b = f >> 5, ch = f & 31;
        sm_ring[b * 48 * C1 + 32 * C1 + ch] = (_Float16)0.f;
      }
    }
    __syncthreads();   // (1) landing buffer + xg visible to all

    // --- convert pass: f32 landing buffer -> f16 operand buffer (edge-masked) ---
    if (c < NCH) {
      const int b = tid >> 4, j = tid & 15;
      for (int rr = j; rr < 320; rr += 16) {
        const int r = rr >> 4;
        const int t = tA - 2 + r;
        const int q = b * 320 + rr;
        h4 hv = {(_Float16)0.f, (_Float16)0.f, (_Float16)0.f, (_Float16)0.f};
        if ((unsigned)t < (unsigned)Tt) {
          const f32x4 v = *reinterpret_cast<const f32x4*>(sm_xef32 + q * 4);
          hv[0] = (_Float16)v[0]; hv[1] = (_Float16)v[1];
          hv[2] = (_Float16)v[2]; hv[3] = (_Float16)v[3];
        }
        *reinterpret_cast<h4*>(sm_xe + q * 4) = hv;
      }
    }
    __syncthreads();   // (2) convert done; landing buffer free for next TDM

    // --- kick TDM for chunk c+1 (overlaps with WMMA below) ---
    if (kTDM && c + 1 < NCH && wid == 0) {
      const int startt = (c + 1) * CHK - 2;   // >= 14 here, no leading pad needed
      tdm_issue_xeeg(x_eeg + (size_t)(b0 * Tt + startt) * CE, Tt - startt);
    }

    // --- conv1 -> BN1 -> ReLU -> ring: shared-A, dual accumulators, sw-pipelined ---
    if (c < NCH) {
#pragma unroll
      for (int bb = 0; bb < 2; ++bb) {
        const int b = 2 * wid + bb;
        const _Float16* xe_b = sm_xe + b * 20 * CE;
        v8f acc0 = {}, acc1 = {};
        v16h a_c  = load_a16(xe_b, CE, 0, lane);
        v16h b0_c = load_bfrag(sm_wk1 + 0 * FRAG_H, lane);
        v16h b1_c = load_bfrag(sm_wk1 + 1 * FRAG_H, lane);
#pragma unroll
        for (int s = 0; s < 10; ++s) {          // s = w*2 + ks
          v16h a_n, b0_n, b1_n;
          if (s + 1 < 10) {                      // prefetch next k-step's fragments
            const int w = (s + 1) >> 1, ks = (s + 1) & 1;
            a_n  = load_a16(xe_b + w * CE, CE, ks * 32, lane);
            b0_n = load_bfrag(sm_wk1 + ((s + 1) * 2 + 0) * FRAG_H, lane);
            b1_n = load_bfrag(sm_wk1 + ((s + 1) * 2 + 1) * FRAG_H, lane);
          }
          acc0 = __builtin_amdgcn_wmma_f32_16x16x32_f16(false, a_c, false, b0_c, (short)0, acc0, false, false);
          acc1 = __builtin_amdgcn_wmma_f32_16x16x32_f16(false, a_c, false, b1_c, (short)0, acc1, false, false);
          a_c = a_n; b0_c = b0_n; b1_c = b1_n;
        }
        const int n = lane & 15, g = lane >> 4;
        const float A0 = sm_bn1A[n],      B0 = sm_bn1B[n];
        const float A1 = sm_bn1A[16 + n], B1 = sm_bn1B[16 + n];
#pragma unroll
        for (int i = 0; i < 8; ++i) {
          const int slot = baseA + i + g * 8;          // <= 47: no modulo needed
          float y0 = acc0[i] * A0 + B0;
          float y1 = acc1[i] * A1 + B1;
          _Float16* row = sm_ring + b * 48 * C1 + slot * C1;
          row[n]      = (_Float16)(y0 > 0.f ? y0 : 0.f);
          row[16 + n] = (_Float16)(y1 > 0.f ? y1 : 0.f);
        }
      }
    }

    // --- conv2 -> BN2 -> ReLU and geo gate (lagging chunk); fragments preloaded ---
    if (c >= 1) {
#pragma unroll
      for (int bb = 0; bb < 2; ++bb) {
        const int b = 2 * wid + bb;
        const _Float16* ring_b = sm_ring + b * 48 * C1;
        v16h a0 = load_a_ring(ring_b, baseB - 1, lane);
        v16h a1 = load_a_ring(ring_b, baseB,     lane);
        v16h a2 = load_a_ring(ring_b, baseB + 1, lane);
        v16h f0 = load_bfrag(sm_wk2 + 0 * FRAG_H, lane);
        v16h f1 = load_bfrag(sm_wk2 + 1 * FRAG_H, lane);
        v16h f2 = load_bfrag(sm_wk2 + 2 * FRAG_H, lane);
        v16h ga = load_a16(sm_xg + b * CHK * 32, 32, 0, lane);
        v16h gb = load_bfrag(sm_gw, lane);
        v8f acc = {};
        acc = __builtin_amdgcn_wmma_f32_16x16x32_f16(false, a0, false, f0, (short)0, acc, false, false);
        acc = __builtin_amdgcn_wmma_f32_16x16x32_f16(false, a1, false, f1, (short)0, acc, false, false);
        acc = __builtin_amdgcn_wmma_f32_16x16x32_f16(false, a2, false, f2, (short)0, acc, false, false);
        v8f gac = {};
        gac = __builtin_amdgcn_wmma_f32_16x16x32_f16(false, ga, false, gb, (short)0, gac, false, false);
        const int n = lane & 15, g = lane >> 4;
        const float A2 = sm_bn2A[n], B2c = sm_bn2B[n];
        const float gbias = sm_geob[n];
#pragma unroll
        for (int i = 0; i < 8; ++i) {
          const int m = i + g * 8;
          float y = acc[i] * A2 + B2c;
          sm_enc[(b * C2 + n) * CHK + m] = y > 0.f ? y : 0.f;           // [b][h][t]
          const float z = gac[i] + gbias;
          sm_geo[(b * C2 + n) * CHK + m] = 1.f / (1.f + __expf(-z));    // [b][h][t]
        }
      }
    }
    __syncthreads();   // (3)

    // --- ALIF scan: 16 sequential steps per (b, hid) thread; contiguous LDS reads ---
    if (c >= 1) {
      const float* ep = sm_enc + (sb * C2 + sh) * CHK;
      const float* gp = sm_geo + (sb * C2 + sh) * CHK;
#pragma unroll
      for (int m = 0; m < CHK; ++m) {
        const int t = tB + m;
        const float e  = ep[m];
        const float gg = gp[m];
        eta = TAU_A * eta + (1.f - TAU_A) * spk;
        const float theta = THETA0 + BETA * eta - gam * gg;
        mem = TAU_M * mem + e;
        spk = (mem - theta) >= 0.f ? 1.f : 0.f;
        mem = mem * (1.f - spk);
        li = 0.9f * li + spk;
        if ((t & 31) < 16) s0 += li; else s1 += li;
        if ((t & 31) == 31) {
          sm_dp[sb * 160 + sh * 10 + (t >> 5)] = (_Float16)((s1 - s0) * 0.0625f);
          s0 = 0.f; s1 = 0.f;
        }
      }
    }
    __syncthreads();   // (4)

    baseA = (baseA == 32) ? 0 : baseA + CHK;   // advance tA % 48
  }

  // ---- FC1: (16 x 160) @ (160 x 64), waves 0..3 take one 16-col tile each ----
  if (wid < 4) {
    v8f acc = {};
#pragma unroll
    for (int ks = 0; ks < 5; ++ks) {
      v16h a  = load_a16(sm_dp, 160, ks * 32, lane);
      v16h bf = load_bfrag(sm_w1 + (ks * 4 + wid) * FRAG_H, lane);
      acc = __builtin_amdgcn_wmma_f32_16x16x32_f16(false, a, false, bf, (short)0, acc, false, false);
    }
    const int n = lane & 15, g = lane >> 4;
    const int col = wid * 16 + n;
#pragma unroll
    for (int i = 0; i < 8; ++i) {
      const int m = i + g * 8;
      const float y = acc[i] + sm_fc1b[col];
      sm_h1[m * 64 + col] = (_Float16)(y > 0.f ? y : 0.f);
    }
  }
  __syncthreads();

  // ---- FC2: (16 x 64) @ (64 x 4 padded to 16), wave 0 ----
  if (wid == 0) {
    v8f acc = {};
#pragma unroll
    for (int ks = 0; ks < 2; ++ks) {
      v16h a  = load_a16(sm_h1, 64, ks * 32, lane);
      v16h bf = load_bfrag(sm_w2 + ks * FRAG_H, lane);
      acc = __builtin_amdgcn_wmma_f32_16x16x32_f16(false, a, false, bf, (short)0, acc, false, false);
    }
    const int n = lane & 15, g = lane >> 4;
    if (n < 4) {
#pragma unroll
      for (int i = 0; i < 8; ++i) {
        const int m = i + g * 8;
        out[(size_t)(b0 + m) * 4 + n] = acc[i] + b2[n];
      }
    }
  }
}

extern "C" void kernel_launch(void* const* d_in, const int* in_sizes, int n_in,
                              void* d_out, int out_size, void* d_ws, size_t ws_size,
                              hipStream_t stream) {
  (void)in_sizes; (void)n_in; (void)out_size; (void)d_ws; (void)ws_size;
  const float* x_eeg = (const float*)d_in[0];
  const float* x_geo = (const float*)d_in[1];
  const float* k1    = (const float*)d_in[2];
  const float* cb1   = (const float*)d_in[3];
  const float* bn1s  = (const float*)d_in[4];
  const float* bn1b  = (const float*)d_in[5];
  const float* bn1m  = (const float*)d_in[6];
  const float* bn1v  = (const float*)d_in[7];
  const float* k2    = (const float*)d_in[8];
  const float* cb2   = (const float*)d_in[9];
  const float* bn2s  = (const float*)d_in[10];
  const float* bn2b  = (const float*)d_in[11];
  const float* bn2m  = (const float*)d_in[12];
  const float* bn2v  = (const float*)d_in[13];
  const float* geoW  = (const float*)d_in[14];
  const float* geob  = (const float*)d_in[15];
  const float* gam   = (const float*)d_in[16];
  const float* w1    = (const float*)d_in[17];
  const float* b1    = (const float*)d_in[18];
  const float* w2    = (const float*)d_in[19];
  const float* b2    = (const float*)d_in[20];
  float* outp = (float*)d_out;

  geo_alif_fused<<<dim3(Bt / BPB), dim3(NTHREADS), SMEM_BYTES, stream>>>(
      x_eeg, x_geo, k1, cb1, bn1s, bn1b, bn1m, bn1v,
      k2, cb2, bn2s, bn2b, bn2m, bn2v, geoW, geob,
      gam, w1, b1, w2, b2, outp);
}
